// W8A8IntLinear_42889543418086
// MI455X (gfx1250) — 8-bit matrix operand order repaired and checked by reading, compiled, not run on hardware
//
#include <hip/hip_runtime.h>

typedef __attribute__((ext_vector_type(8))) int   v8i;
typedef __attribute__((ext_vector_type(4))) int   v4i;
typedef __attribute__((ext_vector_type(2))) int   v2i;

__device__ __forceinline__ void async_ld16(unsigned lds_addr, unsigned g_off,
                                           const void* g_base) {
    asm volatile("global_load_async_to_lds_b128 %0, %1, %2"
                 :: "v"(lds_addr), "v"(g_off), "s"(g_base)
                 : "memory");
}
__device__ __forceinline__ void wait_async0() {
    asm volatile("s_wait_asynccnt 0x0" ::: "memory");
}

__global__ __launch_bounds__(256) void quant_kernel(
    const float* __restrict__ x, signed char* __restrict__ q,
    float* __restrict__ row_scale, int* __restrict__ row_azp, int K)
{
    const int row  = blockIdx.x;
    const int base = threadIdx.x * 16;
    const float* xr = x + (size_t)row * K;

    float vals[16];
#pragma unroll
    for (int i = 0; i < 16; i += 4) {
        const float4 v = *(const float4*)(xr + base + i);
        vals[i + 0] = v.x; vals[i + 1] = v.y;
        vals[i + 2] = v.z; vals[i + 3] = v.w;
    }
    float vmax = vals[0], vmin = vals[0];
#pragma unroll
    for (int i = 1; i < 16; ++i) {
        vmax = fmaxf(vmax, vals[i]);
        vmin = fminf(vmin, vals[i]);
    }
#pragma unroll
    for (int off = 16; off > 0; off >>= 1) {
        vmax = fmaxf(vmax, __shfl_xor(vmax, off, 32));
        vmin = fminf(vmin, __shfl_xor(vmin, off, 32));
    }
    __shared__ float smax[8], smin[8];
    const int wave = threadIdx.x >> 5, lane = threadIdx.x & 31;
    if (lane == 0) { smax[wave] = vmax; smin[wave] = vmin; }
    __syncthreads();
    vmax = smax[0]; vmin = smin[0];
#pragma unroll
    for (int w = 1; w < 8; ++w) {
        vmax = fmaxf(vmax, smax[w]);
        vmin = fminf(vmin, smin[w]);
    }

    const float scale = (vmax - vmin) * (1.0f / 255.0f);
    const float inv   = 1.0f / scale;
    const int   azp   = __float2int_rn(-128.0f - vmin * inv);
    if (threadIdx.x == 0) { row_scale[row] = scale; row_azp[row] = azp; }

    int packed[4];
#pragma unroll
    for (int i = 0; i < 16; ++i) {
        int qi = __float2int_rn(vals[i] * inv) + azp;
        qi = min(127, max(-128, qi));
        ((signed char*)packed)[i] = (signed char)qi;
    }
    *(int4*)(q + (size_t)row * K + base) = *(const int4*)packed;
}

__global__ __launch_bounds__(256) void wprep_kernel(
    const int* __restrict__ w, signed char* __restrict__ wq,
    int* __restrict__ adj, int K)
{
    const int n    = blockIdx.x;
    const int base = threadIdx.x * 16;
    const int* wr  = w + (size_t)n * K;

    int sum = 0;
    int packed[4];
#pragma unroll
    for (int i = 0; i < 16; ++i) {
        const int v = wr[base + i];
        sum += v;
        ((signed char*)packed)[i] = (signed char)v;
    }
    *(int4*)(wq + (size_t)n * K + base) = *(const int4*)packed;

#pragma unroll
    for (int off = 16; off > 0; off >>= 1) sum += __shfl_xor(sum, off, 32);
    __shared__ int ssum[8];
    const int wave = threadIdx.x >> 5, lane = threadIdx.x & 31;
    if (lane == 0) ssum[wave] = sum;
    __syncthreads();
    if (threadIdx.x == 0) {
        int t = 0;
#pragma unroll
        for (int i = 0; i < 8; ++i) t += ssum[i];
        adj[n] = t;
    }
}

#define LDS_TILE 8192
#define BUF_STRIDE (2 * LDS_TILE)

__global__ __launch_bounds__(256) void gemm_kernel(
    const signed char* __restrict__ qa,
    const signed char* __restrict__ qb,
    const float* __restrict__ row_scale,
    const int*   __restrict__ row_azp,
    const int*   __restrict__ adj,
    const float* __restrict__ wscale,
    const float* __restrict__ bias,
    float* __restrict__ out,
    int K, int Nn)
{
    __shared__ __align__(16) signed char smem[2 * BUF_STRIDE];

    const int tid    = threadIdx.x;
    const int wave   = tid >> 5;
    const int lane   = tid & 31;
    const int l      = lane & 15;
    const int hi     = lane >> 4;
    const int wave_m = wave & 3;
    const int wave_n = wave >> 2;

    const int m0 = blockIdx.y * 128;
    const int n0 = blockIdx.x * 128;

    const int lr = tid >> 2;
    const int lc = (tid & 3) << 4;

    const unsigned lds_base = (unsigned)(size_t)(&smem[0]);
    const unsigned stA      = lds_base + (unsigned)(lr * 64 + lc);
    const unsigned gA       = (unsigned)((m0 + lr) * K + lc);
    const unsigned gB       = (unsigned)((n0 + lr) * K + lc);
    const unsigned rowStep  = (unsigned)(64 * K);

    v8i acc[2][4];
    const v8i vzero = {0, 0, 0, 0, 0, 0, 0, 0};
#pragma unroll
    for (int mt = 0; mt < 2; ++mt)
#pragma unroll
        for (int nt = 0; nt < 4; ++nt) acc[mt][nt] = vzero;

    auto issue = [&](unsigned bufOfs, int kstep) {
        const unsigned kb = (unsigned)(kstep << 6);
        const unsigned la = stA + bufOfs;
        const unsigned lb = la + LDS_TILE;
        async_ld16(la,         gA + kb,           qa);
        async_ld16(la + 4096u, gA + kb + rowStep, qa);
        async_ld16(lb,         gB + kb,           qb);
        async_ld16(lb + 4096u, gB + kb + rowStep, qb);
    };

    auto compute = [&](unsigned bufOfs) {
        const signed char* curA = smem + bufOfs;
        const signed char* curB = curA + LDS_TILE;

        v8i Af[2];
#pragma unroll
        for (int mt = 0; mt < 2; ++mt) {
            const signed char* ab = curA + (wave_m * 32 + mt * 16 + l) * 64 + hi * 16;
            const v2i p0 = *(const v2i*)(ab + 0);
            const v2i p1 = *(const v2i*)(ab + 8);
            const v2i p2 = *(const v2i*)(ab + 32);
            const v2i p3 = *(const v2i*)(ab + 40);
            v8i a;
            a[0] = p0[0]; a[1] = p0[1]; a[2] = p1[0]; a[3] = p1[1];
            a[4] = p2[0]; a[5] = p2[1]; a[6] = p3[0]; a[7] = p3[1];
            Af[mt] = a;
        }
        v8i Bf[4];
#pragma unroll
        for (int nt = 0; nt < 4; ++nt) {
            const signed char* bb = curB + (wave_n * 64 + nt * 16 + l) * 64 + hi * 16;
            const v4i q0 = *(const v4i*)(bb + 0);
            const v4i q1 = *(const v4i*)(bb + 32);
            v8i b;
            b[0] = q0[0]; b[1] = q0[1]; b[2] = q0[2]; b[3] = q0[3];
            b[4] = q1[0]; b[5] = q1[1]; b[6] = q1[2]; b[7] = q1[3];
            Bf[nt] = b;
        }
#pragma unroll
        for (int mt = 0; mt < 2; ++mt)
#pragma unroll
            for (int nt = 0; nt < 4; ++nt)
                acc[mt][nt] = __builtin_amdgcn_wmma_i32_16x16x64_iu8(
                    true, Af[mt], true, Bf[nt], acc[mt][nt], false, false);
    };

    const int ksteps = K >> 6;

    issue(0u, 0);

    int kt = 0;
    for (; kt + 2 < ksteps; kt += 2) {
        wait_async0();
        __syncthreads();
        issue(BUF_STRIDE, kt + 1);
        compute(0u);

        wait_async0();
        __syncthreads();
        issue(0u, kt + 2);
        compute(BUF_STRIDE);
    }

    wait_async0();
    __syncthreads();
    issue(BUF_STRIDE, kt + 1);
    compute(0u);

    wait_async0();
    __syncthreads();
    compute(BUF_STRIDE);

    const int m_wave = m0 + wave_m * 32;
    const int n_wave = n0 + wave_n * 64;

#pragma unroll
    for (int mt = 0; mt < 2; ++mt) {
        const int mrow = m_wave + mt * 16 + hi * 8;
        float sc[8]; int az[8];
        *(float4*)(sc + 0) = *(const float4*)(row_scale + mrow);
        *(float4*)(sc + 4) = *(const float4*)(row_scale + mrow + 4);
        *(int4*)(az + 0)   = *(const int4*)(row_azp + mrow);
        *(int4*)(az + 4)   = *(const int4*)(row_azp + mrow + 4);
#pragma unroll
        for (int nt = 0; nt < 4; ++nt) {
            const int n  = n_wave + nt * 16 + l;
            const float ws = wscale[n];
            const float bv = bias[n];
            const int   ad = adj[n];
            const v8i   a  = acc[mt][nt];
#pragma unroll
            for (int v = 0; v < 8; ++v) {
                const int corrected = a[v] - az[v] * ad;
                out[(size_t)(mrow + v) * Nn + n] = sc[v] * ws * (float)corrected + bv;
            }
        }
    }
}

extern "C" void kernel_launch(void* const* d_in, const int* in_sizes, int n_in,
                              void* d_out, int out_size, void* d_ws, size_t ws_size,
                              hipStream_t stream)
{
    const float* x      = (const float*)d_in[0];
    const int*   w      = (const int*)d_in[1];
    const float* wscale = (const float*)d_in[2];
    const float* bias   = (const float*)d_in[3];
    float*       out    = (float*)d_out;

    const int Nn = in_sizes[2];
    const int K  = in_sizes[1] / Nn;
    const int Mm = in_sizes[0] / K;

    char* ws = (char*)d_ws;
    size_t o = 0;
    signed char* q      = (signed char*)(ws + o); o += (size_t)Mm * K;
    signed char* wq     = (signed char*)(ws + o); o += (size_t)Nn * K;
    float*       rscale = (float*)(ws + o);       o += (size_t)Mm * sizeof(float);
    int*         razp   = (int*)(ws + o);         o += (size_t)Mm * sizeof(int);
    int*         adj    = (int*)(ws + o);

    quant_kernel<<<Mm, 256, 0, stream>>>(x, q, rscale, razp, K);
    wprep_kernel<<<Nn, 256, 0, stream>>>(w, wq, adj, K);

    dim3 grid(Nn / 128, Mm / 128);
    gemm_kernel<<<grid, 256, 0, stream>>>(q, wq, rscale, razp, adj, wscale, bias,
                                          out, K, Nn);
}
